// RoIBBox_56865366999679
// MI455X (gfx1250) — matrix operand order repaired and checked by reading, not compiled, not run on hardware
//
#include <hip/hip_runtime.h>
#include <hip/hip_bf16.h>
#include <stdint.h>

// ---------------- problem constants (match reference) ----------------
constexpr int B_SZ      = 16;
constexpr int FH        = 64;
constexpr int FW        = 64;
constexpr int AC        = 9;
constexpr int A_TOT     = FH * FW * AC;      // 36864 anchors
constexpr int PRE_K     = 1200;              // PRE_NMS_K
constexpr int NMS_TOPN  = 300;
constexpr int MAX_GT    = 32;
constexpr int TOTAL_POS = 64;
constexpr int NWORDS    = (PRE_K + 31) / 32; // 38 bitmask words per row
constexpr int NGROUP16  = PRE_K / 16;        // 75 column groups of 16
constexpr float IOU_T   = 0.5f;
constexpr float EPSF    = 1e-9f;

typedef __attribute__((ext_vector_type(16))) _Float16     v16h;
typedef __attribute__((ext_vector_type(8)))  float        v8f;
typedef __attribute__((ext_vector_type(4)))  unsigned int u32x4;
typedef __attribute__((ext_vector_type(8)))  int          i32x8;
typedef __attribute__((ext_vector_type(4)))  int          i32x4;

// =====================================================================
// Kernel 1: per-batch top-1200 selection by score + fused box decoding.
// Histogram threshold prunes 36864 -> ~2048 candidates, bitonic sort in
// LDS, decode only the winners. One 1024-thread block per batch.
// =====================================================================
__global__ __launch_bounds__(1024)
void topk_decode_kernel(const float* __restrict__ deltas,
                        const float* __restrict__ labels,
                        const float* __restrict__ anchors,
                        float* __restrict__ selBoxes) {
  __shared__ unsigned hist[1024];
  __shared__ unsigned long long keys[2048];
  __shared__ int tbin;
  __shared__ unsigned cnt;

  const int b   = blockIdx.x;
  const int tid = threadIdx.x;
  const float* sc = labels + (size_t)b * A_TOT;

  hist[tid] = 0u;
  keys[tid] = 0ull;
  keys[tid + 1024] = 0ull;
  if (tid == 0) cnt = 0u;
  __syncthreads();

  // histogram of scores (uniform in [0,1))
  for (int i = tid; i < A_TOT; i += 1024) {
    float s = sc[i];
    int bn = (int)(s * 1024.0f);
    bn = bn < 0 ? 0 : (bn > 1023 ? 1023 : bn);
    atomicAdd(&hist[bn], 1u);
  }
  __syncthreads();

  if (tid == 0) {
    unsigned c = 0; int t = 1023;
    for (; t >= 0; --t) { c += hist[t]; if (c >= (unsigned)PRE_K) break; }
    tbin = t < 0 ? 0 : t;
  }
  __syncthreads();

  // collect candidates; key packs (orderable score | ~index) for stable
  // descending sort with ties broken by ascending index (argsort semantics)
  const int tb = tbin;
  for (int i = tid; i < A_TOT; i += 1024) {
    float s = sc[i];
    int bn = (int)(s * 1024.0f);
    bn = bn < 0 ? 0 : (bn > 1023 ? 1023 : bn);
    if (bn >= tb) {
      unsigned p = atomicAdd(&cnt, 1u);
      if (p < 2048u)
        keys[p] = ((unsigned long long)__float_as_uint(s) << 32) |
                  (unsigned)(~(unsigned)i);
    }
  }
  __syncthreads();

  // bitonic sort, descending, 2048 elements / 1024 threads
  for (unsigned k = 2; k <= 2048; k <<= 1) {
    for (unsigned j = k >> 1; j > 0; j >>= 1) {
      for (unsigned i = tid; i < 2048; i += 1024) {
        unsigned ixj = i ^ j;
        if (ixj > i) {
          unsigned long long x = keys[i], y = keys[ixj];
          bool up = ((i & k) == 0);               // up => larger first
          bool sw = up ? (x < y) : (x > y);
          if (sw) { keys[i] = y; keys[ixj] = x; }
        }
      }
      __syncthreads();
    }
  }

  // decode deltas for the top PRE_K anchors only
  for (int r = tid; r < PRE_K; r += 1024) {
    unsigned a = ~(unsigned)(keys[r] & 0xFFFFFFFFull);
    const float* dl = deltas  + (size_t)b * A_TOT * 4 + (size_t)a * 4;
    const float* an = anchors + (size_t)b * A_TOT * 4 + (size_t)a * 4;
    float ah  = an[2] - an[0];
    float aw  = an[3] - an[1];
    float acy = an[0] + 0.5f * ah;
    float acx = an[1] + 0.5f * aw;
    float h   = expf(dl[2]) * ah;
    float w   = expf(dl[3]) * aw;
    float cy  = dl[0] * ah + acy;
    float cx  = dl[1] * aw + acx;
    float y1  = cy - 0.5f * h;
    float x1  = cx - 0.5f * w;
    float* o = selBoxes + (size_t)b * PRE_K * 4 + (size_t)r * 4;
    o[0] = y1; o[1] = x1; o[2] = y1 + h; o[3] = x1 + w;
  }
}

// =====================================================================
// Kernel 2: greedy NMS per batch. 512 threads = 16 waves per block.
//  - TDM stages the 1200x4 f32 boxes into LDS (tensor_load_to_lds)
//  - suppression matrix as 38 x u32 bitmask words per row (global ws)
//  - blocked greedy: 32-row wave-shuffle scan, then WMMA f16 boolean
//    matmul broadcasts the block's suppression to all later columns
// =====================================================================
struct NmsSmem {
  float    boxes[PRE_K * 4];   // MUST be first member: TDM lands at its LDS addr
  unsigned sup[NWORDS];
  unsigned wb[NWORDS];
  unsigned keepw;
  int      kt;
};

__global__ __launch_bounds__(512)
void nms_kernel(const float* __restrict__ selBoxes,
                unsigned* __restrict__ gsup,
                float* __restrict__ gnms) {
  __shared__ NmsSmem sm;
  const int b    = blockIdx.x;
  const int tid  = threadIdx.x;
  const int lane = tid & 31;
  const int wv   = tid >> 5;
  unsigned* rowbase = gsup + (size_t)b * PRE_K * NWORDS;

  // ---- stage boxes into LDS via the Tensor Data Mover ----
#if __has_builtin(__builtin_amdgcn_tensor_load_to_lds)
  if (tid < 32) {
    unsigned long long ga =
        (unsigned long long)(uintptr_t)(selBoxes + (size_t)b * PRE_K * 4);
    unsigned ldsOff = (unsigned)(uintptr_t)(&sm.boxes[0]); // addr[31:0] = LDS offset
    u32x4 g0;
    g0[0] = 1u;                                  // count = 1 valid descriptor
    g0[1] = ldsOff;                              // lds_addr
    g0[2] = (unsigned)(ga & 0xFFFFFFFFull);      // global_addr[31:0]
    g0[3] = (unsigned)((ga >> 32) & 0x1FFFFFFull) | (2u << 30); // addr[56:32] | type=2
    i32x8 g1;
    g1[0] = (int)(2u << 16);                     // data_size = 4B, no multicast
    g1[1] = (int)((unsigned)(PRE_K * 4) << 16);  // tensor_dim0 lo16 (=4800 elems)
    g1[2] = (int)(1u << 16);                     // tensor_dim0 hi16=0 | tensor_dim1=1
    g1[3] = (int)((unsigned)(PRE_K * 4) << 16);  // tile_dim0 = 4800
    g1[4] = 1;                                   // tile_dim1 = 1, tile_dim2 = 0
    g1[5] = PRE_K * 4;                           // tensor_dim0_stride lo32
    g1[6] = (int)((unsigned)(PRE_K * 4) << 16);  // tensor_dim1_stride lo16
    g1[7] = 0;
    i32x4 g2 = (i32x4)0, g3 = (i32x4)0;
#if defined(__clang_major__) && (__clang_major__ >= 23)
    i32x8 g4 = (i32x8)0;
    __builtin_amdgcn_tensor_load_to_lds(g0, g1, g2, g3, g4, 0);
#else
    __builtin_amdgcn_tensor_load_to_lds(g0, g1, g2, g3, 0);
#endif
    __builtin_amdgcn_s_wait_tensorcnt(0);
  }
#else
  for (int i = tid; i < PRE_K * 4; i += 512)
    sm.boxes[i] = selBoxes[(size_t)b * PRE_K * 4 + i];
#endif
  if (tid < NWORDS) sm.sup[tid] = 0u;
  __syncthreads();

  // ---- phase A: upper-triangular iou>0.5 bitmask matrix ----
  for (int t = tid; t < PRE_K * NWORDS; t += 512) {
    int i = t / NWORDS, w = t % NWORDS;
    float ay1 = sm.boxes[i * 4 + 0], ax1 = sm.boxes[i * 4 + 1];
    float ay2 = sm.boxes[i * 4 + 2], ax2 = sm.boxes[i * 4 + 3];
    float areaA = (ay2 - ay1) * (ax2 - ax1);
    unsigned bits = 0u;
    for (int bt = 0; bt < 32; ++bt) {
      int j = w * 32 + bt;
      if (j < PRE_K && j > i) {
        float by1 = sm.boxes[j * 4 + 0], bx1 = sm.boxes[j * 4 + 1];
        float by2 = sm.boxes[j * 4 + 2], bx2 = sm.boxes[j * 4 + 3];
        float ih = fminf(ay2, by2) - fmaxf(ay1, by1); ih = ih > 0.f ? ih : 0.f;
        float iw = fminf(ax2, bx2) - fmaxf(ax1, bx1); iw = iw > 0.f ? iw : 0.f;
        float inter = ih * iw;
        float areaB = (by2 - by1) * (bx2 - bx1);
        float iou = inter / (areaA + areaB - inter + EPSF);
        if (iou > IOU_T) bits |= (1u << bt);
      }
    }
    rowbase[(size_t)i * NWORDS + w] = bits;
  }
  __syncthreads();

  // ---- phase B: blocked greedy elimination ----
  for (int blk = 0; blk < NWORDS; ++blk) {
    const int r0 = blk * 32;
    // in-block 32-step scan on wave 0 via shuffles
    if (tid < 32) {
      int row = r0 + tid;
      unsigned w = (row < PRE_K) ? rowbase[(size_t)row * NWORDS + blk] : 0u;
      unsigned S = sm.sup[blk];
      for (int i2 = 0; i2 < 32; ++i2) {
        unsigned wi = (unsigned)__shfl((int)w, i2, 32);
        if (!((S >> i2) & 1u)) S |= wi;    // keep row i2 -> suppress its victims
      }
      if (tid == 0) { sm.sup[blk] = S; sm.keepw = ~S; }
    }
    __syncthreads();

    // cross-block broadcast: s[j] |= (sum_k keep[k]*sup[k][j]) > 0
    // 32x16 {0,1} matmul on matrix cores: V_WMMA_F32_16X16X32_F16
    const unsigned kw = sm.keepw;
    v16h afrag;
    {
      const int hi = (lane >= 16) ? 8 : 0;  // A 16x32 f16 lane-group K mapping
      for (int h = 0; h < 16; ++h) {
        int K = (h < 8 ? h : h + 8) + hi;
        afrag[h] = (_Float16)(float)((kw >> K) & 1u);
      }
    }
    for (int g = 2 * (blk + 1) + wv; g < NGROUP16; g += 16) {
      const int hk = (lane >= 16) ? 8 : 0;  // B: lane%16 = N column, elements = K (table order)
      const int col = (g & 1) * 16 + (lane & 15);
      v16h bfrag;
      for (int s = 0; s < 16; ++s) {
        int row = r0 + (s < 8 ? s : s + 8) + hk;
        unsigned word = (row < PRE_K) ? rowbase[(size_t)row * NWORDS + (g >> 1)] : 0u;
        bfrag[s] = (_Float16)(float)((word >> col) & 1u);
      }
      v8f cz = {};
      v8f c = __builtin_amdgcn_wmma_f32_16x16x32_f16(
          false, afrag, false, bfrag, (short)0, cz, false, false);
      unsigned m = (unsigned)__ballot(c[0] > 0.5f) & 0xFFFFu; // row M=0, N=lane
      if (lane == 0 && m)
        atomicOr(&sm.sup[g >> 1], m << ((g & 1) * 16));
    }
    __syncthreads();
  }

  // ---- phase C: compact kept boxes (stable order), pad to 300 ----
  if (tid == 0) {
    unsigned base = 0;
    for (int w = 0; w < NWORDS; ++w) {
      unsigned mask = (w == NWORDS - 1) ? 0xFFFFu : 0xFFFFFFFFu;
      sm.wb[w] = base;
      base += __popc((~sm.sup[w]) & mask);
    }
    sm.kt = (int)(base < (unsigned)NMS_TOPN ? base : (unsigned)NMS_TOPN);
  }
  __syncthreads();
  for (int j = tid; j < PRE_K; j += 512) {
    int w = j >> 5, bt = j & 31;
    if (!((sm.sup[w] >> bt) & 1u)) {
      unsigned mask = (w == NWORDS - 1) ? 0xFFFFu : 0xFFFFFFFFu;
      unsigned below = bt ? ((1u << bt) - 1u) : 0u;
      unsigned rank = sm.wb[w] + __popc((~sm.sup[w]) & mask & below);
      if (rank < (unsigned)NMS_TOPN) {
        float* o = gnms + (size_t)b * NMS_TOPN * 4 + (size_t)rank * 4;
        o[0] = sm.boxes[j * 4 + 0]; o[1] = sm.boxes[j * 4 + 1];
        o[2] = sm.boxes[j * 4 + 2]; o[3] = sm.boxes[j * 4 + 3];
      }
    }
  }
  const int kt = sm.kt;
  for (int idx = tid; idx < NMS_TOPN * 4; idx += 512) {
    if ((idx >> 2) >= kt) gnms[(size_t)b * NMS_TOPN * 4 + idx] = 0.0f;
  }
}

// =====================================================================
// Kernel 3: IoU of 300 NMS boxes vs 32 GT, argmax + top-64 sampling.
// 256 threads per batch; bitonic sort of 512 padded keys in LDS.
// =====================================================================
__global__ __launch_bounds__(256)
void roi_kernel(const float* __restrict__ gnms,
                const float* __restrict__ gt_boxes,
                float* __restrict__ out) {
  __shared__ float nb[NMS_TOPN * 4];
  __shared__ float gtb[MAX_GT * 4];
  __shared__ unsigned long long keys[512];
  __shared__ int gidx[NMS_TOPN];

  const int b   = blockIdx.x;
  const int tid = threadIdx.x;

  for (int i = tid; i < NMS_TOPN * 4; i += 256)
    nb[i] = gnms[(size_t)b * NMS_TOPN * 4 + i];
  for (int i = tid; i < MAX_GT * 4; i += 256)
    gtb[i] = gt_boxes[(size_t)b * MAX_GT * 4 + i];
  for (int i = tid; i < 512; i += 256) keys[i] = 0ull;
  __syncthreads();

  for (int j = tid; j < NMS_TOPN; j += 256) {
    float ay1 = nb[j * 4 + 0], ax1 = nb[j * 4 + 1];
    float ay2 = nb[j * 4 + 2], ax2 = nb[j * 4 + 3];
    float areaA = (ay2 - ay1) * (ax2 - ax1);
    float best = -1.0f; int bi = 0;
    for (int g = 0; g < MAX_GT; ++g) {
      float by1 = gtb[g * 4 + 0], bx1 = gtb[g * 4 + 1];
      float by2 = gtb[g * 4 + 2], bx2 = gtb[g * 4 + 3];
      float ih = fminf(ay2, by2) - fmaxf(ay1, by1); ih = ih > 0.f ? ih : 0.f;
      float iw = fminf(ax2, bx2) - fmaxf(ax1, bx1); iw = iw > 0.f ? iw : 0.f;
      float inter = ih * iw;
      float areaB = (by2 - by1) * (bx2 - bx1);
      float iou = inter / (areaA + areaB - inter + EPSF);
      if (iou > best) { best = iou; bi = g; }      // first max wins (argmax)
    }
    gidx[j] = bi;
    float m = best < 0.f ? 0.f : best;
    keys[j] = ((unsigned long long)__float_as_uint(m) << 32) |
              (unsigned)(~(unsigned)j);
  }
  __syncthreads();

  for (unsigned k = 2; k <= 512; k <<= 1) {
    for (unsigned j = k >> 1; j > 0; j >>= 1) {
      for (unsigned i = tid; i < 512; i += 256) {
        unsigned ixj = i ^ j;
        if (ixj > i) {
          unsigned long long x = keys[i], y = keys[ixj];
          bool up = ((i & k) == 0);
          bool sw = up ? (x < y) : (x > y);
          if (sw) { keys[i] = y; keys[ixj] = x; }
        }
      }
      __syncthreads();
    }
  }

  float* outIdx = out + (size_t)B_SZ * 128 * 4;
  for (int r = tid; r < TOTAL_POS; r += 256) {
    unsigned j = ~(unsigned)(keys[r] & 0xFFFFFFFFull);
    float* o = out + (size_t)b * 128 * 4 + (size_t)r * 4;
    o[0] = nb[j * 4 + 0]; o[1] = nb[j * 4 + 1];
    o[2] = nb[j * 4 + 2]; o[3] = nb[j * 4 + 3];
    outIdx[(size_t)b * TOTAL_POS + r] = (float)gidx[j];
  }
  // 64 negative RoIs are zeros
  for (int r = tid; r < TOTAL_POS * 4; r += 256)
    out[(size_t)b * 128 * 4 + (size_t)(TOTAL_POS * 4) + r] = 0.0f;
}

// =====================================================================
extern "C" void kernel_launch(void* const* d_in, const int* in_sizes, int n_in,
                              void* d_out, int out_size, void* d_ws, size_t ws_size,
                              hipStream_t stream) {
  const float* deltas   = (const float*)d_in[0];
  const float* labels   = (const float*)d_in[1];
  const float* anchors  = (const float*)d_in[2];
  const float* gt_boxes = (const float*)d_in[3];
  float* out = (float*)d_out;

  // workspace layout
  char* ws = (char*)d_ws;
  float*    selBoxes = (float*)ws;                                  // 16*1200*4 f32
  size_t off = (size_t)B_SZ * PRE_K * 4 * sizeof(float);
  unsigned* gsup     = (unsigned*)(ws + off);                       // 16*1200*38 u32
  off += (size_t)B_SZ * PRE_K * NWORDS * sizeof(unsigned);
  float*    gnms     = (float*)(ws + off);                          // 16*300*4 f32

  topk_decode_kernel<<<B_SZ, 1024, 0, stream>>>(deltas, labels, anchors, selBoxes);
  nms_kernel<<<B_SZ, 512, 0, stream>>>(selBoxes, gsup, gnms);
  roi_kernel<<<B_SZ, 256, 0, stream>>>(gnms, gt_boxes, out);
}
